// FastAttentionBlock_7000796692776
// MI455X (gfx1250) — compile-verified, matrix operand order checked by reading, not run on hardware
//
#include <hip/hip_runtime.h>
#include <hip/hip_bf16.h>
#include <math.h>

// ---------------------------------------------------------------------------
// FastAttentionBlock on MI455X (gfx1250, wave32, WMMA f32_16x16x32_f16).
// f16 A/B fragments, f32 accumulation. LDS tiles in fragment-ready permuted
// K order (fragment = two aligned ds_load_b128). Pure-f16 tile fills use the
// CDNA5 async global->LDS path (ASYNCcnt) when the builtin is available.
// ---------------------------------------------------------------------------

typedef __attribute__((ext_vector_type(16))) _Float16 v16h;
typedef __attribute__((ext_vector_type(8)))  _Float16 v8h;
typedef __attribute__((ext_vector_type(2)))  _Float16 v2h;
typedef __attribute__((ext_vector_type(8)))  float    v8f;
typedef int v4i_ __attribute__((vector_size(16)));

#define NPOS 4096   // 64*64 positions per batch
#define CIN  512
#define CK   64
#define CSMF 128
#define NB   8
#define FP   40     // LDS fragment pitch (halves): 80B, keeps 16B alignment

#if __has_builtin(__builtin_amdgcn_global_load_async_to_lds_b128)
#define ASYNC_CP 1
#else
#define ASYNC_CP 0
#endif

// 16B global -> LDS copy; async (no VGPR round-trip) when available.
__device__ __forceinline__ void cp16(void* lds, const void* g) {
#if ASYNC_CP
  v4i_* gp = (v4i_*)g;   // strip const, generic pointer of the right type
  v4i_* lp = (v4i_*)lds;
  __builtin_amdgcn_global_load_async_to_lds_b128(
      (__attribute__((address_space(1))) v4i_*)gp,
      (__attribute__((address_space(3))) v4i_*)lp, 0, 0);
#else
  *(v8h*)lds = *(const v8h*)g;
#endif
}

__device__ __forceinline__ void async_wait() {
#if ASYNC_CP
#if __has_builtin(__builtin_amdgcn_s_wait_asynccnt)
  __builtin_amdgcn_s_wait_asynccnt(0);
#else
  asm volatile("s_wait_asynccnt 0x0" ::: "memory");
#endif
#endif
}

// Storage slot for K index k (0..31) so that lane-half h reads 16 contiguous
// halves at offset 16*h:  half h needs K in {8h..8h+7} U {16+8h..16+8h+7}.
__device__ __forceinline__ int sperm(int k) {
  return ((k & 8) ? 16 : 0) + ((k & 16) ? 8 : 0) + (k & 7);
}

// Load one 16-half fragment (32B) as two aligned 16B LDS loads.
__device__ __forceinline__ v16h frag_ld(const _Float16* p) {
  v8h a = *(const v8h*)p;
  v8h b = *(const v8h*)(p + 8);
  return __builtin_shufflevector(a, b, 0, 1, 2, 3, 4, 5, 6, 7, 8, 9, 10, 11,
                                 12, 13, 14, 15);
}

__device__ __forceinline__ v8f wmma32(v16h a, v16h b, v8f c) {
  return __builtin_amdgcn_wmma_f32_16x16x32_f16(false, a, false, b, (short)0, c,
                                                false, false);
}

// ---------------------------------------------------------------------------
// Kernel 1: fused q/k/v 1x1 conv GEMM.  Y[oc,p] = sum_c W[oc,c] * X[c,p].
// q: affine + per-position L2 norm -> qn [B][P][64]   (position-major)
// k: affine + per-position L2 norm -> kn [B][64][P]   (channel-major)
// v: affine + ReLU                 -> vh [B][512][P]  (channel-major)
// grid: (32 p-tiles of 128, 10 [0=q,1=k,2..9=v], 8 batches); 256 thr = 8 waves
// ---------------------------------------------------------------------------
__global__ __launch_bounds__(256) void qkv_gemm_kernel(
    const float* __restrict__ x,
    const float* __restrict__ wq, const float* __restrict__ sq, const float* __restrict__ bq,
    const float* __restrict__ wk, const float* __restrict__ sk, const float* __restrict__ bk,
    const float* __restrict__ wv, const float* __restrict__ sv, const float* __restrict__ bv,
    _Float16* __restrict__ qn, _Float16* __restrict__ kn, _Float16* __restrict__ vh) {
  __shared__ _Float16 Af[64][FP];
  __shared__ _Float16 Bf[128][FP];
  __shared__ float    red[4][128];

  const int tid = threadIdx.x;
  const int lane = tid & 31, wave = tid >> 5;
  const int half = lane >> 4, lo = lane & 15;
  const int wm = wave & 3, wn = wave >> 2;
  const int b = blockIdx.z;
  const int pbase = blockIdx.x * 128;
  const int mode = blockIdx.y;

  const float* w;
  const float* sc;
  const float* bi;
  if (mode == 0)      { w = wq; sc = sq; bi = bq; }
  else if (mode == 1) { w = wk; sc = sk; bi = bk; }
  else {
    int oc0 = (mode - 2) * 64;
    w = wv + (size_t)oc0 * CIN; sc = sv + oc0; bi = bv + oc0;
  }
  const float* xb = x + (size_t)b * CIN * NPOS;

  v8f acc[4] = {};

  for (int kk = 0; kk < CIN; kk += 32) {
    if (kk + 32 < CIN)
      __builtin_prefetch(&xb[(size_t)(kk + 32) * NPOS + pbase], 0, 1);
    // A: 64 oc x 32 k weights (f32 -> f16 pairs)
#pragma unroll
    for (int i = 0; i < 4; ++i) {
      int t = tid + i * 256;
      int r = t >> 4, kp = (t & 15) * 2;
      float2 f = *(const float2*)&w[(size_t)r * CIN + kk + kp];
      v2h h; h.x = (_Float16)f.x; h.y = (_Float16)f.y;
      *(v2h*)&Af[r][sperm(kp)] = h;
    }
    // B: 32 k x 128 p activations, float4 over positions, packed k-pairs
#pragma unroll
    for (int i = 0; i < 2; ++i) {
      int t = tid + i * 256;
      int kp = (t >> 5) * 2;
      int c4 = (t & 31) * 4;
      const float* src = &xb[(size_t)(kk + kp) * NPOS + pbase + c4];
      float4 f0 = *(const float4*)src;
      float4 f1 = *(const float4*)(src + NPOS);
      int slot = sperm(kp);
      v2h h;
      h.x = (_Float16)f0.x; h.y = (_Float16)f1.x; *(v2h*)&Bf[c4 + 0][slot] = h;
      h.x = (_Float16)f0.y; h.y = (_Float16)f1.y; *(v2h*)&Bf[c4 + 1][slot] = h;
      h.x = (_Float16)f0.z; h.y = (_Float16)f1.z; *(v2h*)&Bf[c4 + 2][slot] = h;
      h.x = (_Float16)f0.w; h.y = (_Float16)f1.w; *(v2h*)&Bf[c4 + 3][slot] = h;
    }
    __syncthreads();
    v16h a = frag_ld(&Af[16 * wm + lo][16 * half]);
    v16h bfr[4];
#pragma unroll
    for (int nb = 0; nb < 4; ++nb)
      bfr[nb] = frag_ld(&Bf[64 * wn + 16 * nb + lo][16 * half]);
#pragma unroll
    for (int nb = 0; nb < 4; ++nb) acc[nb] = wmma32(a, bfr[nb], acc[nb]);
    __syncthreads();
  }

  // affine; C layout: element r -> (M = r + 8*half, N = lo)
  float vals[4][8];
#pragma unroll
  for (int nb = 0; nb < 4; ++nb)
#pragma unroll
    for (int r = 0; r < 8; ++r) {
      int ch = 16 * wm + r + 8 * half;
      vals[nb][r] = acc[nb][r] * sc[ch] + bi[ch];
    }

  if (mode >= 2) {                         // v: ReLU, channel-major store
    int oc0 = (mode - 2) * 64;
#pragma unroll
    for (int nb = 0; nb < 4; ++nb)
#pragma unroll
      for (int r = 0; r < 8; ++r) {
        int ch = 16 * wm + r + 8 * half;
        int col = 64 * wn + 16 * nb + lo;
        vh[((size_t)b * CIN + oc0 + ch) * NPOS + pbase + col] =
            (_Float16)fmaxf(vals[nb][r], 0.f);
      }
  } else {                                 // q/k: per-position L2 norm (64 ch)
#pragma unroll
    for (int nb = 0; nb < 4; ++nb) {
      float ssq = 0.f;
#pragma unroll
      for (int r = 0; r < 8; ++r) ssq += vals[nb][r] * vals[nb][r];
      ssq += __shfl_xor(ssq, 16, 32);
      if (half == 0) red[wm][64 * wn + 16 * nb + lo] = ssq;
    }
    __syncthreads();
#pragma unroll
    for (int nb = 0; nb < 4; ++nb) {
      int col = 64 * wn + 16 * nb + lo;
      float tot = red[0][col] + red[1][col] + red[2][col] + red[3][col];
      float inv = 1.f / fmaxf(sqrtf(tot), 1e-12f);
      if (mode == 0) {                     // qn: position-major, 16B store
        v8h hv;
#pragma unroll
        for (int r = 0; r < 8; ++r) hv[r] = (_Float16)(vals[nb][r] * inv);
        *(v8h*)&qn[((size_t)b * NPOS + pbase + col) * CK + 16 * wm + 8 * half] = hv;
      } else {                             // kn: channel-major (K contiguous
#pragma unroll                             //     along N for sim_kernel's A)
        for (int r = 0; r < 8; ++r) {
          int ch = 16 * wm + r + 8 * half;
          kn[((size_t)b * CK + ch) * NPOS + pbase + col] =
              (_Float16)(vals[nb][r] * inv);
        }
      }
    }
  }
}

// ---------------------------------------------------------------------------
// Kernel 2: sim[m,c] = sum_n kn[m,n] * v[c,n]  (K = 4096), output transposed
// as simT [B][512][64]. Fills are async global->LDS 16B copies.
// grid: (8 c-tiles of 64, 1, 8 batches); each wave 16x32 (2 accumulators)
// ---------------------------------------------------------------------------
__global__ __launch_bounds__(256) void sim_kernel(
    const _Float16* __restrict__ kn, const _Float16* __restrict__ vh,
    _Float16* __restrict__ simT) {
  __shared__ _Float16 Af[64][FP];
  __shared__ _Float16 Bf[64][FP];

  const int tid = threadIdx.x;
  const int lane = tid & 31, wave = tid >> 5;
  const int half = lane >> 4, lo = lane & 15;
  const int wm = wave & 3, wn = wave >> 2;
  const int b = blockIdx.z;
  const int cbase = blockIdx.x * 64;

  const int fr = tid >> 2, fg = (tid & 3) * 8;   // fill row/col + k-group
  const int fslot = sperm(fg);

  v8f acc[2] = {};
  for (int n0 = 0; n0 < NPOS; n0 += 32) {
    cp16(&Af[fr][fslot], &kn[((size_t)b * CK + fr) * NPOS + n0 + fg]);
    cp16(&Bf[fr][fslot], &vh[((size_t)b * CIN + cbase + fr) * NPOS + n0 + fg]);
    async_wait();
    __syncthreads();
    v16h a = frag_ld(&Af[16 * wm + lo][16 * half]);
    v16h bfr[2];
#pragma unroll
    for (int nb = 0; nb < 2; ++nb)
      bfr[nb] = frag_ld(&Bf[32 * wn + 16 * nb + lo][16 * half]);
#pragma unroll
    for (int nb = 0; nb < 2; ++nb) acc[nb] = wmma32(a, bfr[nb], acc[nb]);
    __syncthreads();
  }
#pragma unroll
  for (int nb = 0; nb < 2; ++nb)
#pragma unroll
    for (int r = 0; r < 8; ++r) {
      int m = 16 * wm + r + 8 * half;
      int c = cbase + 32 * wn + 16 * nb + lo;
      simT[((size_t)b * CIN + c) * CK + m] = (_Float16)acc[nb][r];
    }
}

// ---------------------------------------------------------------------------
// Kernel 3: ctx[p,c] = sum_m sim[m,c] * qn[p,m]  (K = 64, LDS-resident)
// output ctxp [B][P][512] position-major. Fills are async 16B copies.
// grid: (32 p-tiles of 128, 8 c-tiles of 64, 8 batches)
// ---------------------------------------------------------------------------
__global__ __launch_bounds__(256) void ctx_kernel(
    const _Float16* __restrict__ simT, const _Float16* __restrict__ qn,
    _Float16* __restrict__ ctxp) {
  __shared__ _Float16 Af[64][2 * FP];
  __shared__ _Float16 Bf[128][2 * FP];

  const int tid = threadIdx.x;
  const int lane = tid & 31, wave = tid >> 5;
  const int half = lane >> 4, lo = lane & 15;
  const int wm = wave & 3, wn = wave >> 2;
  const int b = blockIdx.z;
  const int cbase = blockIdx.y * 64;
  const int pbase = blockIdx.x * 128;

#pragma unroll
  for (int i = 0; i < 2; ++i) {            // A: simT 64c x 64m
    int t = tid + i * 256;
    int c = t >> 3, g = (t & 7) * 8;
    int slot = (g >> 5) * FP + sperm(g & 31);
    cp16(&Af[c][slot], &simT[((size_t)b * CIN + cbase + c) * CK + g]);
  }
#pragma unroll
  for (int i = 0; i < 4; ++i) {            // B: qn 128p x 64m
    int t = tid + i * 256;
    int p = t >> 3, g = (t & 7) * 8;
    int slot = (g >> 5) * FP + sperm(g & 31);
    cp16(&Bf[p][slot], &qn[((size_t)b * NPOS + pbase + p) * CK + g]);
  }
  async_wait();
  __syncthreads();

  v8f acc[4] = {};
#pragma unroll
  for (int kt = 0; kt < 2; ++kt) {
    v16h a = frag_ld(&Af[16 * wm + lo][kt * FP + 16 * half]);
    v16h bfr[4];
#pragma unroll
    for (int nb = 0; nb < 4; ++nb)
      bfr[nb] = frag_ld(&Bf[64 * wn + 16 * nb + lo][kt * FP + 16 * half]);
#pragma unroll
    for (int nb = 0; nb < 4; ++nb) acc[nb] = wmma32(a, bfr[nb], acc[nb]);
  }
#pragma unroll
  for (int nb = 0; nb < 4; ++nb) {
    int col = pbase + 64 * wn + 16 * nb + lo;
    v8h hv;
#pragma unroll
    for (int r = 0; r < 8; ++r) hv[r] = (_Float16)acc[nb][r];
    *(v8h*)&ctxp[((size_t)b * NPOS + col) * CIN + cbase + 16 * wm + 8 * half] = hv;
  }
}

// ---------------------------------------------------------------------------
// Kernel 4: o = wo . ctx ; fuse = relu(o*so+bo) + x + up
// output fusep [B][P][512] position-major. B fill is async 16B copies.
// grid: (32 p-tiles, 8 oc-tiles of 64, 8 batches)
// ---------------------------------------------------------------------------
__global__ __launch_bounds__(256) void oproj_fuse_kernel(
    const _Float16* __restrict__ ctxp, const float* __restrict__ wo,
    const float* __restrict__ so, const float* __restrict__ bo,
    const float* __restrict__ x, const float* __restrict__ up,
    _Float16* __restrict__ fusep) {
  __shared__ _Float16 Af[64][FP];
  __shared__ _Float16 Bf[128][FP];

  const int tid = threadIdx.x;
  const int lane = tid & 31, wave = tid >> 5;
  const int half = lane >> 4, lo = lane & 15;
  const int wm = wave & 3, wn = wave >> 2;
  const int b = blockIdx.z;
  const int ocb = blockIdx.y * 64;
  const int pbase = blockIdx.x * 128;

  v8f acc[4] = {};
  for (int kk = 0; kk < CIN; kk += 32) {
#pragma unroll
    for (int i = 0; i < 2; ++i) {          // B: ctxp 128p x 32c (async)
      int t = tid + i * 256;
      int p = t >> 2, g = (t & 3) * 8;
      cp16(&Bf[p][sperm(g)],
           &ctxp[((size_t)b * NPOS + pbase + p) * CIN + kk + g]);
    }
#pragma unroll
    for (int i = 0; i < 4; ++i) {          // A: wo 64oc x 32c (f32 -> f16)
      int t = tid + i * 256;
      int r = t >> 4, kp = (t & 15) * 2;
      float2 f = *(const float2*)&wo[(size_t)(ocb + r) * CIN + kk + kp];
      v2h h; h.x = (_Float16)f.x; h.y = (_Float16)f.y;
      *(v2h*)&Af[r][sperm(kp)] = h;
    }
    async_wait();
    __syncthreads();
    v16h a = frag_ld(&Af[16 * wm + lo][16 * half]);
    v16h bfr[4];
#pragma unroll
    for (int nb = 0; nb < 4; ++nb)
      bfr[nb] = frag_ld(&Bf[64 * wn + 16 * nb + lo][16 * half]);
#pragma unroll
    for (int nb = 0; nb < 4; ++nb) acc[nb] = wmma32(a, bfr[nb], acc[nb]);
    __syncthreads();
  }
#pragma unroll
  for (int nb = 0; nb < 4; ++nb) {
    int col = pbase + 64 * wn + 16 * nb + lo;
    v8h hv;
#pragma unroll
    for (int r = 0; r < 8; ++r) {
      int oc = ocb + 16 * wm + r + 8 * half;
      size_t gi = ((size_t)b * CIN + oc) * NPOS + col;
      hv[r] = (_Float16)(fmaxf(acc[nb][r] * so[oc] + bo[oc], 0.f) +
                         x[gi] + up[gi]);
    }
    *(v8h*)&fusep[((size_t)b * NPOS + col) * CIN + ocb + 16 * wm + 8 * half] = hv;
  }
}

// ---------------------------------------------------------------------------
// Kernel 5: 3x3 conv (pad 1) implicit GEMM over K = 512*9, affine + ReLU.
// B operand from position-major fusep: async 16B copies in the interior,
// direct zero stores on the halo. grid: (32 p-tiles, 2 oc-tiles, 8 batches)
// ---------------------------------------------------------------------------
__global__ __launch_bounds__(256) void smooth_conv3_kernel(
    const _Float16* __restrict__ fusep, const float* __restrict__ ws,
    const float* __restrict__ ss, const float* __restrict__ bs,
    float* __restrict__ out) {
  __shared__ _Float16 Af[64][FP];
  __shared__ _Float16 Bf[128][FP];

  const int tid = threadIdx.x;
  const int lane = tid & 31, wave = tid >> 5;
  const int half = lane >> 4, lo = lane & 15;
  const int wm = wave & 3, wn = wave >> 2;
  const int b = blockIdx.z;
  const int ocb = blockIdx.y * 64;
  const int pbase = blockIdx.x * 128;
  const int h0 = pbase >> 6;               // this block covers rows h0, h0+1

  v8f acc[4] = {};
  for (int t9 = 0; t9 < 9; ++t9) {
    int ky = t9 / 3 - 1, kx = t9 % 3 - 1;
    for (int icb = 0; icb < CIN; icb += 32) {
#pragma unroll
      for (int i = 0; i < 2; ++i) {        // B: shifted fusep, zero halo
        int t = tid + i * 256;
        int p = t >> 2, g = (t & 3) * 8;
        int hh = h0 + (p >> 6) + ky;
        int wwp = (p & 63) + kx;
        if ((unsigned)hh < 64u && (unsigned)wwp < 64u) {
          cp16(&Bf[p][sperm(g)],
               &fusep[((size_t)b * NPOS + hh * 64 + wwp) * CIN + icb + g]);
        } else {
          v8h z = {};
          *(v8h*)&Bf[p][sperm(g)] = z;
        }
      }
#pragma unroll
      for (int i = 0; i < 8; ++i) {        // A: ws[oc][ic][ky][kx] (stride 9)
        int t = tid + i * 256;
        int r = t >> 5, c = t & 31;
        Af[r][sperm(c)] = (_Float16)
            ws[((size_t)(ocb + r) * CIN + icb + c) * 9 + (size_t)t9];
      }
      async_wait();
      __syncthreads();
      v16h a = frag_ld(&Af[16 * wm + lo][16 * half]);
      v16h bfr[4];
#pragma unroll
      for (int nb = 0; nb < 4; ++nb)
        bfr[nb] = frag_ld(&Bf[64 * wn + 16 * nb + lo][16 * half]);
#pragma unroll
      for (int nb = 0; nb < 4; ++nb) acc[nb] = wmma32(a, bfr[nb], acc[nb]);
      __syncthreads();
    }
  }
#pragma unroll
  for (int nb = 0; nb < 4; ++nb)
#pragma unroll
    for (int r = 0; r < 8; ++r) {
      int oc = ocb + 16 * wm + r + 8 * half;
      int p = pbase + 64 * wn + 16 * nb + lo;
      out[((size_t)b * CSMF + oc) * NPOS + p] =
          fmaxf(acc[nb][r] * ss[oc] + bs[oc], 0.f);
    }
}

// ---------------------------------------------------------------------------
extern "C" void kernel_launch(void* const* d_in, const int* in_sizes, int n_in,
                              void* d_out, int out_size, void* d_ws,
                              size_t ws_size, hipStream_t stream) {
  (void)in_sizes; (void)n_in; (void)out_size; (void)ws_size;
  const float* x   = (const float*)d_in[0];
  const float* up  = (const float*)d_in[1];
  const float* wq  = (const float*)d_in[2];
  const float* sq  = (const float*)d_in[3];
  const float* bq  = (const float*)d_in[4];
  const float* wk  = (const float*)d_in[5];
  const float* sk  = (const float*)d_in[6];
  const float* bk  = (const float*)d_in[7];
  const float* wv  = (const float*)d_in[8];
  const float* sv  = (const float*)d_in[9];
  const float* bv  = (const float*)d_in[10];
  const float* wo  = (const float*)d_in[11];
  const float* so  = (const float*)d_in[12];
  const float* bo  = (const float*)d_in[13];
  const float* wsw = (const float*)d_in[14];
  const float* ssc = (const float*)d_in[15];
  const float* bsc = (const float*)d_in[16];
  float* out = (float*)d_out;

  char* wsb = (char*)d_ws;
  size_t off = 0;
  auto wsalloc = [&](size_t bytes) -> void* {
    void* p = wsb + off;
    off += (bytes + 255) & ~(size_t)255;
    return p;
  };
  _Float16* qn    = (_Float16*)wsalloc((size_t)NB * NPOS * CK  * 2); // [B][P][64]
  _Float16* kn    = (_Float16*)wsalloc((size_t)NB * CK  * NPOS * 2); // [B][64][P]
  _Float16* vh    = (_Float16*)wsalloc((size_t)NB * CIN * NPOS * 2); // [B][512][P]
  _Float16* simT  = (_Float16*)wsalloc((size_t)NB * CIN * CK   * 2); // [B][512][64]
  _Float16* fusep = (_Float16*)wsalloc((size_t)NB * NPOS * CIN * 2); // [B][P][512]
  _Float16* ctxp  = vh;  // v is dead after sim_kernel; reuse for ctx [B][P][512]

  qkv_gemm_kernel<<<dim3(32, 10, NB), 256, 0, stream>>>(
      x, wq, sq, bq, wk, sk, bk, wv, sv, bv, qn, kn, vh);
  sim_kernel<<<dim3(8, 1, NB), 256, 0, stream>>>(kn, vh, simT);
  ctx_kernel<<<dim3(32, 8, NB), 256, 0, stream>>>(simT, qn, ctxp);
  oproj_fuse_kernel<<<dim3(32, 8, NB), 256, 0, stream>>>(
      ctxp, wo, so, bo, x, up, fusep);
  smooth_conv3_kernel<<<dim3(32, 2, NB), 256, 0, stream>>>(
      fusep, wsw, ssc, bsc, out);
}
